// GraphAttention_65755949301760
// MI455X (gfx1250) — matrix operand order repaired and checked by reading, not compiled, not run on hardware
//
#include <hip/hip_runtime.h>
#include <hip/hip_bf16.h>

#define N_NODES 8192
#define D_IN    512
#define D_OUT   64
#define HEADS   8
#define ALPHA   0.2f
#define NWORDS  (N_NODES / 32)
#define BPITCH  40   // LDS row pitch (bf16 elems) for 32-wide B tile, bank spread

typedef __attribute__((ext_vector_type(16))) __bf16 v16bf;
typedef __attribute__((ext_vector_type(8)))  __bf16 v8bf;
typedef __attribute__((ext_vector_type(8)))  float  v8f;
typedef __attribute__((ext_vector_type(4)))  float  v4f;

union V16 { v16bf v; v8bf h[2]; };

// ---------------------------------------------------------------------------
// Pack adj (int32 0/2) into a bitmask: 256 MB -> 8 MB so the per-head sweep
// reads bits from L2 instead of HBM.
// ---------------------------------------------------------------------------
__global__ __launch_bounds__(256) void gat_adjpack(const int* __restrict__ adj,
                                                   unsigned* __restrict__ bits) {
  size_t w = (size_t)blockIdx.x * blockDim.x + threadIdx.x;  // one 32-col word
  const int* p = adj + w * 32;
  unsigned b = 0;
#pragma unroll
  for (int i = 0; i < 32; ++i) b |= (unsigned)(p[i] > 0) << i;
  bits[w] = b;
}

// ---------------------------------------------------------------------------
// x (f32) -> xbf (bf16), 8 elements per thread.
// ---------------------------------------------------------------------------
__global__ __launch_bounds__(256) void gat_cvt_x(const float* __restrict__ x,
                                                 __bf16* __restrict__ xbf) {
  size_t i = ((size_t)blockIdx.x * blockDim.x + threadIdx.x) * 8;
  v4f a = *(const v4f*)(x + i);
  v4f b = *(const v4f*)(x + i + 4);
  v8bf o;
#pragma unroll
  for (int j = 0; j < 4; ++j) { o[j] = (__bf16)a[j]; o[4 + j] = (__bf16)b[j]; }
  *(v8bf*)(xbf + i) = o;
}

// ---------------------------------------------------------------------------
// W [H][512][64] f32 -> WT [H][64][512] bf16 (contiguous-K B fragments).
// ---------------------------------------------------------------------------
__global__ __launch_bounds__(256) void gat_packW(const float* __restrict__ W,
                                                 __bf16* __restrict__ WT) {
  int idx = blockIdx.x * blockDim.x + threadIdx.x;  // h*64 + c, 512 total
  int h = idx >> 6, c = idx & 63;
  for (int k = 0; k < D_IN; ++k)
    WT[((size_t)h * D_OUT + c) * D_IN + k] =
        (__bf16)W[((size_t)h * D_IN + k) * D_OUT + c];
}

// ---------------------------------------------------------------------------
// Projection: Wh[h] = x @ W[h] via bf16 WMMA. One wave = 16x64 tile of one
// head. All fragment loads are contiguous bf16 (16B / 32B per lane).
// ---------------------------------------------------------------------------
__global__ __launch_bounds__(256) void gat_proj(const __bf16* __restrict__ xbf,
                                                const __bf16* __restrict__ WT,
                                                __bf16* __restrict__ Wh,
                                                __bf16* __restrict__ WhT) {
  int wave = threadIdx.x >> 5, lane = threadIdx.x & 31;
  int wid = blockIdx.x * 8 + wave;       // 4096 waves: 8 heads x 512 row blocks
  int h  = wid >> 9;
  int r0 = (wid & 511) << 4;
  int nlo = lane & 15, half = lane >> 4;

  v8f acc[4] = {};

  for (int k0 = 0; k0 < D_IN; k0 += 32) {
    const __bf16* xr = xbf + (size_t)(r0 + nlo) * D_IN + k0;
    V16 A;                                   // K = {8*half+i, 16+8*half+i}
    A.h[0] = *(const v8bf*)(xr + 8 * half);
    A.h[1] = *(const v8bf*)(xr + 16 + 8 * half);
#pragma unroll
    for (int t = 0; t < 4; ++t) {
      const __bf16* wp = WT + ((size_t)h * D_OUT + t * 16 + nlo) * D_IN
                            + k0;  // K = {8*half+i, 16+8*half+i}
      V16 B;
      B.h[0] = *(const v8bf*)(wp + 8 * half);
      B.h[1] = *(const v8bf*)(wp + 16 + 8 * half);
      acc[t] = __builtin_amdgcn_wmma_f32_16x16x32_bf16(
          false, A.v, false, B.v, (short)0, acc[t], false, false);
    }
  }

  // D layout: (VGPR r, lane l) = (M = r + 8*half, N = nlo)
#pragma unroll
  for (int t = 0; t < 4; ++t) {
    int c = t * 16 + nlo;
    v8bf col;
#pragma unroll
    for (int r = 0; r < 8; ++r) {
      float v = acc[t][r];
      int row = r0 + 8 * half + r;
      Wh[((size_t)h * N_NODES + row) * D_OUT + c] = (__bf16)v;
      col[r] = (__bf16)v;
    }
    *(v8bf*)(WhT + ((size_t)h * D_OUT + c) * N_NODES + r0 + 8 * half) = col;
  }
}

// ---------------------------------------------------------------------------
// f1[h,i] = Wh[h,i,:] . a1[h], f2 likewise.
// ---------------------------------------------------------------------------
__global__ __launch_bounds__(256) void gat_f1f2(const __bf16* __restrict__ Wh,
                                                const float* __restrict__ a1,
                                                const float* __restrict__ a2,
                                                float* __restrict__ f1,
                                                float* __restrict__ f2) {
  int idx = blockIdx.x * blockDim.x + threadIdx.x;  // h*N + i
  int h = idx >> 13;
  const __bf16* wr = Wh + (size_t)idx * D_OUT;
  float s1 = 0.f, s2 = 0.f;
#pragma unroll
  for (int n = 0; n < D_OUT; ++n) {
    float v = (float)wr[n];
    s1 += v * a1[h * D_OUT + n];
    s2 += v * a2[h * D_OUT + n];
  }
  f1[idx] = s1;
  f2[idx] = s2;
}

// ---------------------------------------------------------------------------
// Per-head max of f2 (analytic softmax max: lrelu(f1[i] + max_j f2[j])).
// ---------------------------------------------------------------------------
__global__ __launch_bounds__(256) void gat_f2max(const float* __restrict__ f2,
                                                 float* __restrict__ f2max) {
  __shared__ float red[8];
  int h = blockIdx.x;
  const float* p = f2 + (size_t)h * N_NODES;
  float m = -3.0e38f;
  for (int i = threadIdx.x; i < N_NODES; i += 256) m = fmaxf(m, p[i]);
#pragma unroll
  for (int s = 16; s > 0; s >>= 1) m = fmaxf(m, __shfl_xor(m, s, 32));
  if ((threadIdx.x & 31) == 0) red[threadIdx.x >> 5] = m;
  __syncthreads();
  if (threadIdx.x == 0) {
    float r = red[0];
#pragma unroll
    for (int w = 1; w < 8; ++w) r = fmaxf(r, red[w]);
    f2max[h] = r;
  }
}

// ---------------------------------------------------------------------------
// Single-sweep masked softmax + (att @ Wh), LDS-blocked.
//   Block  = 8 waves -> 256 rows of one head.
//   Wave   = 32 rows (two 16-row A fragments share every B fragment).
//   B tile = WhT[h][0..63][j0..j0+31] (4 KB), cooperatively staged in LDS,
//            double-buffered, one __syncthreads per 32-column chunk.
// B traffic drops 16x vs per-wave streaming (4 GB -> 256 MB from L2).
// Softmax max is analytic (lrelu(f1 + max f2)); sums ride along with the
// unnormalized WMMA accumulation; 1/sum applied at store via shuffles.
// ---------------------------------------------------------------------------
__global__ __launch_bounds__(256) void gat_attn(const unsigned* __restrict__ adjbits,
                                                const __bf16* __restrict__ WhT,
                                                const float* __restrict__ f1,
                                                const float* __restrict__ f2,
                                                const float* __restrict__ f2max,
                                                float* __restrict__ out) {
  __shared__ __bf16 Bld[2][D_OUT * BPITCH];   // 2 x 5 KB

  int wave = threadIdx.x >> 5, lane = threadIdx.x & 31;
  int h  = blockIdx.x >> 5;                   // 32 blocks per head
  int R0 = (blockIdx.x & 31) << 8;            // 256 rows per block
  int r0 = R0 + wave * 32;                    // 32 rows per wave
  int nlo = lane & 15, half = lane >> 4;
  int rowA = r0 + nlo, rowB = r0 + 16 + nlo;

  const float* f2h = f2 + (size_t)h * N_NODES;
  const __bf16* WhTh = WhT + (size_t)h * D_OUT * N_NODES;
  const unsigned* arowA = adjbits + (size_t)rowA * NWORDS;
  const unsigned* arowB = adjbits + (size_t)rowB * NWORDS;

  float f1a = f1[(size_t)h * N_NODES + rowA];
  float f1b = f1[(size_t)h * N_NODES + rowB];
  float fm = f2max[h];
  float mxa = f1a + fm; mxa = mxa > 0.f ? mxa : ALPHA * mxa;  // exact row max
  float mxb = f1b + fm; mxb = mxb > 0.f ? mxb : ALPHA * mxb;

  // cooperative fill coordinates: thread -> (c, part) loads 8 bf16 (16 B)
  int fc = threadIdx.x >> 2;                  // 0..63
  int fp = (threadIdx.x & 3) * 8;             // 0,8,16,24
  const __bf16* fsrc = WhTh + (size_t)fc * N_NODES + fp;

  // prologue: stage chunk 0 into buffer 0
  *(v8bf*)&Bld[0][fc * BPITCH + fp] = *(const v8bf*)fsrc;

  float lsa = 0.f, lsb = 0.f;
  v8f accA[4] = {};
  v8f accB[4] = {};

  for (int jc = 0; jc < N_NODES / 32; ++jc) {
    int j0 = jc * 32;
    __syncthreads();  // buf[jc&1] ready; all reads of buf[(jc+1)&1] retired
    if (jc + 1 < N_NODES / 32)
      *(v8bf*)&Bld[(jc + 1) & 1][fc * BPITCH + fp] =
          *(const v8bf*)(fsrc + j0 + 32);

    unsigned wa = arowA[jc], wb = arowB[jc];
    unsigned aLo = wa >> (8 * half), aHi = wa >> (16 + 8 * half);
    unsigned bLo = wb >> (8 * half), bHi = wb >> (16 + 8 * half);

    v4f fva = *(const v4f*)(f2h + j0 + 8 * half);
    v4f fvb = *(const v4f*)(f2h + j0 + 8 * half + 4);
    v4f fvc = *(const v4f*)(f2h + j0 + 16 + 8 * half);
    v4f fvd = *(const v4f*)(f2h + j0 + 16 + 8 * half + 4);
    float fv[16];
#pragma unroll
    for (int i = 0; i < 4; ++i) {
      fv[i] = fva[i]; fv[4 + i] = fvb[i]; fv[8 + i] = fvc[i]; fv[12 + i] = fvd[i];
    }

    V16 Aa, Ab;  // K = {8*half+i, 16+8*half+i} per the A-fragment layout
#pragma unroll
    for (int i = 0; i < 16; ++i) {
      unsigned onA = (i < 8) ? ((aLo >> i) & 1u) : ((aHi >> (i - 8)) & 1u);
      unsigned onB = (i < 8) ? ((bLo >> i) & 1u) : ((bHi >> (i - 8)) & 1u);
      float ea = f1a + fv[i]; ea = ea > 0.f ? ea : ALPHA * ea;
      float eb = f1b + fv[i]; eb = eb > 0.f ? eb : ALPHA * eb;
      float pa = onA ? __expf(ea - mxa) : 0.f;
      float pb = onB ? __expf(eb - mxb) : 0.f;
      lsa += pa; lsb += pb;
      Aa.v[i] = (__bf16)pa;
      Ab.v[i] = (__bf16)pb;
    }

    const __bf16* bbuf = &Bld[jc & 1][0];
#pragma unroll
    for (int t = 0; t < 4; ++t) {
      const __bf16* bp = bbuf + (t * 16 + nlo) * BPITCH;
      V16 B;
      B.h[0] = *(const v8bf*)(bp + 8 * half);        // 16 B LDS reads (pitch keeps them
      B.h[1] = *(const v8bf*)(bp + 16 + 8 * half);  //  16 B aligned, banks spread)
      accA[t] = __builtin_amdgcn_wmma_f32_16x16x32_bf16(
          false, Aa.v, false, B.v, (short)0, accA[t], false, false);
      accB[t] = __builtin_amdgcn_wmma_f32_16x16x32_bf16(
          false, Ab.v, false, B.v, (short)0, accB[t], false, false);
    }
  }

  // Row sums: combine half-rows; lane (nlo) holds 1/sum for its two rows.
  lsa += __shfl_xor(lsa, 16, 32);
  lsb += __shfl_xor(lsb, 16, 32);
  float invA = 1.f / fmaxf(lsa, 1e-30f);
  float invB = 1.f / fmaxf(lsb, 1e-30f);

  // Accumulator element (t, r) is row M = r + 8*half; its 1/sum is in lane M.
  float invrA[8], invrB[8];
#pragma unroll
  for (int r = 0; r < 8; ++r) {
    invrA[r] = __shfl(invA, 8 * half + r, 32);
    invrB[r] = __shfl(invB, 8 * half + r, 32);
  }

#pragma unroll
  for (int t = 0; t < 4; ++t) {
#pragma unroll
    for (int r = 0; r < 8; ++r) {
      int ra = r0 + 8 * half + r;
      int rb = ra + 16;
      size_t col = (size_t)h * D_OUT + t * 16 + nlo;
      out[(size_t)ra * (HEADS * D_OUT) + col] = accA[t][r] * invrA[r];
      out[(size_t)rb * (HEADS * D_OUT) + col] = accB[t][r] * invrB[r];
    }
  }
}

// ---------------------------------------------------------------------------
extern "C" void kernel_launch(void* const* d_in, const int* in_sizes, int n_in,
                              void* d_out, int out_size, void* d_ws, size_t ws_size,
                              hipStream_t stream) {
  (void)in_sizes; (void)n_in; (void)out_size; (void)ws_size;
  const float* x  = (const float*)d_in[0];
  const int*   adj = (const int*)d_in[1];
  const float* W  = (const float*)d_in[2];
  const float* a1 = (const float*)d_in[3];
  const float* a2 = (const float*)d_in[4];
  float* out = (float*)d_out;

  char* ws = (char*)d_ws;
  unsigned* adjbits = (unsigned*)ws;                           // 8 MB
  __bf16* Wh  = (__bf16*)(ws + (size_t)(8  << 20));            // 8 MB
  __bf16* WhT = (__bf16*)(ws + (size_t)(16 << 20));            // 8 MB
  float*  f1  = (float*) (ws + (size_t)(24 << 20));            // 256 KB
  float*  f2  = (float*) (ws + (size_t)(24 << 20) + (256 << 10));
  float*  f2m = (float*) (ws + (size_t)(24 << 20) + (512 << 10));
  __bf16* xbf = (__bf16*)(ws + (size_t)(25 << 20));            // 8 MB
  __bf16* WT  = (__bf16*)(ws + (size_t)(33 << 20));            // 512 KB

  gat_adjpack<<<(N_NODES * NWORDS) / 256, 256, 0, stream>>>(adj, adjbits);
  gat_cvt_x<<<(N_NODES * D_IN) / (256 * 8), 256, 0, stream>>>(x, xbf);
  gat_packW<<<(HEADS * D_OUT) / 256, 256, 0, stream>>>(W, WT);
  gat_proj<<<512, 256, 0, stream>>>(xbf, WT, Wh, WhT);
  gat_f1f2<<<(HEADS * N_NODES) / 256, 256, 0, stream>>>(Wh, a1, a2, f1, f2);
  gat_f2max<<<HEADS, 256, 0, stream>>>(f2, f2m);
  gat_attn<<<HEADS * (N_NODES / 256), 256, 0, stream>>>(adjbits, WhT, f1, f2,
                                                        f2m, out);
}
